// SocialModel_72816875537051
// MI455X (gfx1250) — matrix operand order repaired and checked by reading, not compiled, not run on hardware
//
#include <hip/hip_runtime.h>

// Social-LSTM for MI455X (gfx1250): fused social-pooling einsum + tensor
// embedding via bf16 WMMA (f32 accumulate), one workgroup per batch,
// recurrent state resident in LDS. All WMMA B-operands are stored in a
// per-lane swizzled fragment layout ([...][kk][lane][16 halfs]) so every
// fragment load is 2x b128 instead of 16 strided u16 gathers.
// bf16 conversions use native hardware cvt (via __bf16 casts).
//
//   te[n,e] = sum_{m,g} grid[n,m,g] * P[m,g,e],  P[m,g,e] = sum_r h[m,r]*Wt[e,g*128+r]
//
// d_ws layout: WtB_sw bf16 (1MB, swizzled) | Wg_sw bf16 (256KB, swizzled)

#define BB 8
#define TT 8
#define NN 128
#define GG 64
#define RR 128
#define EE 64
#define II 2
#define OO 5
#define PL 12

typedef __attribute__((ext_vector_type(16))) __bf16          v16bf;
typedef __attribute__((ext_vector_type(8)))  __bf16          v8bf;
typedef __attribute__((ext_vector_type(16))) unsigned short  v16u;
typedef __attribute__((ext_vector_type(8)))  unsigned short  v8u;
typedef __attribute__((ext_vector_type(8)))  float           v8f;

__device__ __forceinline__ unsigned short f2bf(float x) {
    __bf16 b = (__bf16)x;                       // native v_cvt f32->bf16
    return __builtin_bit_cast(unsigned short, b);
}
__device__ __forceinline__ float bf2f(unsigned short h) {
    return (float)__builtin_bit_cast(__bf16, h);
}
__device__ __forceinline__ float sigm(float x) { return 1.f / (1.f + __expf(-x)); }
__device__ __forceinline__ float tanh_f(float x) {
    float e = __expf(-2.f * x);
    return (1.f - e) / (1.f + e);
}

// ---- fragment loaders ----------------------------------------------------
// A (16x32 bf16), row-major source with leading dim ld (halfs):
// lane l: M = l&15, K-half offset = (l>>4)*16; VGPRs 0-3 hold K=khalf+0..7,
// VGPRs 4-7 hold K=khalf+8..15 -> two contiguous 16B chunks.
__device__ __forceinline__ v16bf load_a(const unsigned short* base, int ld,
                                        int m0, int k0, int lane) {
    const unsigned short* p = base + (m0 + (lane & 15)) * ld + k0 + ((lane >> 4) << 4);
    union { v16u v; struct { v8u lo, hi; } s; } u;
    u.s.lo = *(const v8u*)p;
    u.s.hi = *(const v8u*)(p + 8);
    return __builtin_bit_cast(v16bf, u.v);
}
// B fragment from swizzled storage: 16 contiguous halfs per lane (32B).
__device__ __forceinline__ v16bf load_bsw(const unsigned short* p, int lane) {
    v16u t = *(const v16u*)(p + lane * 16);
    return __builtin_bit_cast(v16bf, t);
}
// Store a WMMA C tile (v8f) into swizzled-B bf16 layout:
// element (row,col) -> block [colblk=col>>4][kk=row>>5], lane=((row&31)>>4)*16+(col&15),
// half = row&15.  A lane's 8 rows are consecutive halfs -> one b128 store.
__device__ __forceinline__ void store_c_bsw(unsigned short* dst, int cb4,
                                            int m0, int lane, v8f acc) {
    int n    = lane & 15;
    int row0 = m0 + ((lane >> 4) << 3);
    int kk   = row0 >> 5;
    int ld   = (((row0 & 31) >> 4) << 4) + n;
    unsigned short* p = dst + ((cb4 + kk) * 32 + ld) * 16 + (row0 & 15);
    v8bf o = __builtin_convertvector(acc, v8bf);   // packed f32->bf16 cvts
    *(v8u*)p = __builtin_bit_cast(v8u, o);
}

#define WMMA_BF16(A, Bv, C) \
    __builtin_amdgcn_wmma_f32_16x16x32_bf16(false, (A), false, (Bv), (short)0, (C), false, false)

// swizzled-B offset for element (k, col) inside a [nblk][kk][lane][16] buffer
__device__ __forceinline__ int bsw_off(int blkbase, int k, int col) {
    return ((blkbase + (k >> 5)) * 32 + (((k & 31) >> 4) << 4) + (col & 15)) * 16 + (k & 15);
}

// ---- weight repack kernel (f32 -> bf16, swizzled fragment layouts) -------
__global__ void prep_weights(const float* __restrict__ Wt,
                             const float* __restrict__ Wih,
                             const float* __restrict__ Whh,
                             unsigned short* __restrict__ WtB,
                             unsigned short* __restrict__ Wg) {
    int i = blockIdx.x * blockDim.x + threadIdx.x;
    int stride = gridDim.x * blockDim.x;
    // WtB_sw: B operand of P GEMM. Logical B[r][g*64+e] = Wt[e][g*128+r].
    // Block index: (g*4 + e>>4), K = r (128 -> kk 0..3).
    for (int idx = i; idx < GG * EE * RR; idx += stride) {
        int g = idx >> 13, e = (idx >> 7) & 63, r = idx & 127;
        int off = bsw_off((g * 4 + (e >> 4)) * 4, r, e);
        WtB[off] = f2bf(Wt[e * 8192 + g * 128 + r]);
    }
    // Wg_sw: B operand of gates GEMM. Logical B[k][j], j in [0,512), K=256.
    // Block index: (q*8 + jb) with q=j>>7, jb=(j&127)>>4; kk 0..7.
    for (int idx = i; idx < 256 * 512; idx += stride) {
        int j = idx >> 8, k = idx & 255;
        int q = j >> 7, col = j & 127;
        int off = bsw_off((q * 8 + (col >> 4)) * 8, k, col);
        Wg[off] = f2bf(k < 128 ? Wih[j * 128 + k] : Whh[j * 128 + (k - 128)]);
    }
}

// ---- main recurrent kernel: one workgroup per batch ----------------------
__global__ __launch_bounds__(512) void social_lstm(
    const float* __restrict__ obs, const float* __restrict__ grids,
    const float* __restrict__ W_in, const float* __restrict__ b_in,
    const float* __restrict__ b_tensor,
    const float* __restrict__ W_out, const float* __restrict__ b_out,
    const float* __restrict__ b_ih, const float* __restrict__ b_hh,
    const unsigned short* __restrict__ WtB, const unsigned short* __restrict__ Wg,
    float* __restrict__ preds) {
    extern __shared__ char smem[];
    unsigned short* ite   = (unsigned short*)smem;          // [128][128] bf16: [ie|te]
    unsigned short* hb0   = ite + NN * RR;                  // h double buffer (bf16)
    unsigned short* hb1   = hb0 + NN * RR;
    float*          cst   = (float*)(hb1 + NN * RR);        // c state f32 [128][128]
    unsigned short* glds0 = (unsigned short*)(cst + NN * RR); // grid slice g0 [n][m]
    unsigned short* glds1 = glds0 + NN * NN;                // grid slice g0+1
    unsigned short* plds  = glds1 + NN * NN;                // P slices, swizzled-B, 32KB
    float*          posl  = (float*)(plds + NN * RR);       // pred positions [128][2]

    const int tid = threadIdx.x, lane = tid & 31, wid = tid >> 5;
    const int b = blockIdx.x;

    for (int i = tid; i < NN * RR; i += 512) { hb0[i] = 0; cst[i] = 0.f; }
    __syncthreads();

    // wave tile assignment for the 128x64 GEMMs (32 tiles, 2 per wave)
    const int n0a = (wid >> 2) * 16,        e0a = (wid & 3) * 16;
    const int n0b = ((wid + 16) >> 2) * 16, e0b = ((wid + 16) & 3) * 16;

    for (int step = 0; step < TT + PL; ++step) {
        unsigned short* hcur = (step & 1) ? hb1 : hb0;
        unsigned short* hnxt = (step & 1) ? hb0 : hb1;
        const int tg = (step < TT) ? step : (TT - 1);
        const float* gptr = grids + (((size_t)b * TT + tg) * NN) * NN * GG;

        // ---- input embedding: ite[:, 0:64] = relu(pos @ W_in^T + b_in) ----
        for (int i = tid; i < NN * EE; i += 512) {
            int n = i >> 6, e = i & 63;
            float px, py;
            if (step <= TT) {
                int ts = (step < TT) ? step : (TT - 1);
                const float* pp = obs + (((size_t)b * TT + ts) * NN + n) * II;
                px = pp[0]; py = pp[1];
            } else {
                px = posl[n * 2]; py = posl[n * 2 + 1];
            }
            float v = px * W_in[e * II] + py * W_in[e * II + 1] + b_in[e];
            ite[n * RR + e] = f2bf(v > 0.f ? v : 0.f);
        }

        // ---- fused social pooling + tensor embedding ----------------------
        v8f acc0 = {}, acc1 = {};
        for (int cc = 0; cc < GG / 2; ++cc) {
            int g0 = cc * 2;
            __syncthreads();   // glds/plds free to overwrite
            // stage grid[n][m][g0:g0+2] -> deinterleaved bf16 slices
            for (int i = tid; i < (NN * NN) / 2; i += 512) {
                int n = i >> 6, mp = (i & 63) << 1;
                const float* sp = gptr + ((size_t)(n * NN + mp)) * GG + g0;
                float2 va = *(const float2*)sp;         // (m,   g0), (m,   g0+1)
                float2 vb = *(const float2*)(sp + GG);  // (m+1, g0), (m+1, g0+1)
                ((unsigned*)glds0)[(n * NN + mp) >> 1] =
                    ((unsigned)f2bf(vb.x) << 16) | (unsigned)f2bf(va.x);
                ((unsigned*)glds1)[(n * NN + mp) >> 1] =
                    ((unsigned)f2bf(vb.y) << 16) | (unsigned)f2bf(va.y);
            }
            if (cc + 1 < GG / 2)  // global_prefetch_b8 of next slice
                __builtin_prefetch(gptr + (size_t)(tid * 32) * GG + g0 + 2, 0, 1);

            // P_g[m,e] = h(128x128) x WtB[:, g*64+e]  (two g slices)
#pragma unroll
            for (int gi = 0; gi < 2; ++gi) {
                int g = g0 + gi;
                v8f pa = {}, pb = {};
                const unsigned short* wa = WtB + (g * 4 + (e0a >> 4)) * 4 * 32 * 16;
                const unsigned short* wb = WtB + (g * 4 + (e0b >> 4)) * 4 * 32 * 16;
#pragma unroll
                for (int kk = 0; kk < 4; ++kk) {
                    v16bf af = load_a(hcur, RR, n0a, kk * 32, lane);
                    pa = WMMA_BF16(af, load_bsw(wa + kk * 512, lane), pa);
                    v16bf af2 = load_a(hcur, RR, n0b, kk * 32, lane);
                    pb = WMMA_BF16(af2, load_bsw(wb + kk * 512, lane), pb);
                }
                store_c_bsw(plds, (gi * 4 + (e0a >> 4)) * 4, n0a, lane, pa);
                store_c_bsw(plds, (gi * 4 + (e0b >> 4)) * 4, n0b, lane, pb);
            }
            __syncthreads();   // P + grid chunk visible

            // te[n,e] += Grid_g(n,m) x P_g(m,e)
#pragma unroll
            for (int gi = 0; gi < 2; ++gi) {
                const unsigned short* gs = gi ? glds1 : glds0;
                const unsigned short* p0 = plds + (gi * 4 + (e0a >> 4)) * 4 * 32 * 16;
                const unsigned short* p1 = plds + (gi * 4 + (e0b >> 4)) * 4 * 32 * 16;
#pragma unroll
                for (int kk = 0; kk < 4; ++kk) {
                    v16bf ga = load_a(gs, NN, n0a, kk * 32, lane);
                    acc0 = WMMA_BF16(ga, load_bsw(p0 + kk * 512, lane), acc0);
                    v16bf ga2 = load_a(gs, NN, n0b, kk * 32, lane);
                    acc1 = WMMA_BF16(ga2, load_bsw(p1 + kk * 512, lane), acc1);
                }
            }
        }

        // ---- te -> ite[:, 64:128] with bias + relu ------------------------
        {
            int col = lane & 15, rb = (lane >> 4) << 3;
#pragma unroll
            for (int v = 0; v < 8; ++v) {
                float t0 = acc0[v] + b_tensor[e0a + col];
                ite[(n0a + rb + v) * RR + 64 + e0a + col] = f2bf(t0 > 0.f ? t0 : 0.f);
                float t1 = acc1[v] + b_tensor[e0b + col];
                ite[(n0b + rb + v) * RR + 64 + e0b + col] = f2bf(t1 > 0.f ? t1 : 0.f);
            }
        }
        __syncthreads();

        // ---- gates GEMM (K=256: [ie|te|h]) + LSTM pointwise update --------
        for (int s = 0; s < 4; ++s) {
            int task = wid + 16 * s;
            int nb = (task >> 3) * 16, jb16 = task & 7;
            v8f gI = {}, gF = {}, gG = {}, gO = {};
#pragma unroll
            for (int ks = 0; ks < 8; ++ks) {
                int k0 = ks * 32;
                v16bf a = (k0 < 128) ? load_a(ite, RR, nb, k0, lane)
                                     : load_a(hcur, RR, nb, k0 - 128, lane);
                const unsigned short* w0 = Wg + (((0 * 8 + jb16) * 8 + ks) * 32) * 16;
                gI = WMMA_BF16(a, load_bsw(w0, lane), gI);
                const unsigned short* w1 = Wg + (((1 * 8 + jb16) * 8 + ks) * 32) * 16;
                gF = WMMA_BF16(a, load_bsw(w1, lane), gF);
                const unsigned short* w2 = Wg + (((2 * 8 + jb16) * 8 + ks) * 32) * 16;
                gG = WMMA_BF16(a, load_bsw(w2, lane), gG);
                const unsigned short* w3 = Wg + (((3 * 8 + jb16) * 8 + ks) * 32) * 16;
                gO = WMMA_BF16(a, load_bsw(w3, lane), gO);
            }
            int j  = jb16 * 16 + (lane & 15);
            float bi = b_ih[j]       + b_hh[j];
            float bf = b_ih[128 + j] + b_hh[128 + j];
            float bg = b_ih[256 + j] + b_hh[256 + j];
            float bo = b_ih[384 + j] + b_hh[384 + j];
            int rb = (lane >> 4) << 3;
#pragma unroll
            for (int v = 0; v < 8; ++v) {
                int idx = (nb + rb + v) * RR + j;
                float iv = sigm(gI[v] + bi);
                float fv = sigm(gF[v] + bf);
                float gv = tanh_f(gG[v] + bg);
                float ov = sigm(gO[v] + bo);
                float cv = fv * cst[idx] + iv * gv;
                cst[idx]  = cv;
                hnxt[idx] = f2bf(ov * tanh_f(cv));
            }
        }
        __syncthreads();

        // ---- prediction output + position feedback ------------------------
        if (step >= TT) {
            if (tid < NN) {
                int n = tid;
                float o0 = b_out[0], o1 = b_out[1], o2 = b_out[2], o3 = b_out[3], o4 = b_out[4];
                for (int r = 0; r < RR; ++r) {
                    float hv = bf2f(hnxt[n * RR + r]);
                    o0 += hv * W_out[0 * RR + r];
                    o1 += hv * W_out[1 * RR + r];
                    o2 += hv * W_out[2 * RR + r];
                    o3 += hv * W_out[3 * RR + r];
                    o4 += hv * W_out[4 * RR + r];
                }
                int tp = step - TT;
                float* op = preds + ((((size_t)b * PL) + tp) * NN + n) * OO;
                op[0] = o0; op[1] = o1; op[2] = o2; op[3] = o3; op[4] = o4;
                posl[n * 2] = o0; posl[n * 2 + 1] = o1;
            }
            __syncthreads();
        }
    }
}

extern "C" void kernel_launch(void* const* d_in, const int* in_sizes, int n_in,
                              void* d_out, int out_size, void* d_ws, size_t ws_size,
                              hipStream_t stream) {
    const float* obs  = (const float*)d_in[0];
    const float* grds = (const float*)d_in[1];
    const float* W_in = (const float*)d_in[2];
    const float* b_in = (const float*)d_in[3];
    const float* W_t  = (const float*)d_in[4];
    const float* b_t  = (const float*)d_in[5];
    const float* W_o  = (const float*)d_in[6];
    const float* b_o  = (const float*)d_in[7];
    const float* W_ih = (const float*)d_in[8];
    const float* W_hh = (const float*)d_in[9];
    const float* b_ih = (const float*)d_in[10];
    const float* b_hh = (const float*)d_in[11];

    unsigned short* WtB = (unsigned short*)d_ws;        // 64*64*128 bf16 = 1 MB (swizzled)
    unsigned short* Wg  = WtB + GG * EE * RR;           // 256*512  bf16 = 256 KB (swizzled)
    (void)ws_size;  // requires ~1.3 MB workspace

    prep_weights<<<256, 256, 0, stream>>>(W_t, W_ih, W_hh, WtB, Wg);

    size_t smem = (size_t)NN * RR * 2 * 3   // ite + h double buffer (bf16)
                + (size_t)NN * RR * 4       // c state f32
                + (size_t)NN * NN * 2 * 2   // grid chunk bf16 (2 deinterleaved slices)
                + (size_t)NN * RR * 2       // P slices (swizzled-B)
                + (size_t)NN * 2 * 4;       // positions
    social_lstm<<<BB, 512, smem, stream>>>(obs, grds, W_in, b_in, b_t, W_o, b_o,
                                           b_ih, b_hh, WtB, Wg, (float*)d_out);
}
